// STARLayer_87995289960771
// MI455X (gfx1250) — matrix operand order repaired and checked by reading, not compiled, not run on hardware
//
#include <hip/hip_runtime.h>

// ---------------------------------------------------------------------------
// STAR layer, domain-gathered grouped GEMM for gfx1250 (MI455X, wave32, WMMA)
//   h = relu( h @ (sk*dk[dom]) + (sb+db[dom]) )  x3,  gather dom = indicator[b]
// - route each row only through its own domain chain (8x fewer FLOPs)
// - v_wmma_f32_16x16x32_bf16 with f32 accumulation
// - LDS double buffer, K-step 64, register-staged global pipeline
// - whole-slice fragment preload + sched_barrier: one ds_load clause,
//   one s_wait_dscnt, then 8 back-to-back WMMAs per slice
// ---------------------------------------------------------------------------

typedef __attribute__((ext_vector_type(16))) __bf16 v16bf;
typedef __attribute__((ext_vector_type(8)))  __bf16 v8bf;
typedef __attribute__((ext_vector_type(8)))  float  v8f;

#define NDOM   8
#define BATCH  4096
#define D_IN   1024
#define H0     1024
#define H1     512
#define H2     256
#define KS     64          // K-step per LDS slice

// ---------------------------------------------------------------------------
// Metadata: bucket batch rows by domain (padded with -1 to the bucket tail)
// ---------------------------------------------------------------------------
__global__ void init_meta(int* __restrict__ cnt, int* __restrict__ rows, int total_rows) {
    int i = blockIdx.x * blockDim.x + threadIdx.x;
    if (i < NDOM) cnt[i] = 0;
    if (i < total_rows) rows[i] = -1;
}

__global__ void scatter_rows(const int* __restrict__ ind, int* __restrict__ cnt,
                             int* __restrict__ rows) {
    int b = blockIdx.x * blockDim.x + threadIdx.x;
    if (b < BATCH) {
        int d = ind[b];
        int pos = atomicAdd(&cnt[d], 1);
        rows[d * BATCH + pos] = b;
    }
}

// ---------------------------------------------------------------------------
// Weight fusion with tiled transpose: wT[d][o][i] = bf16(sk[i][o]*dk[d][i][o])
// ---------------------------------------------------------------------------
__global__ __launch_bounds__(256) void fuse_weights_t(
    const float* __restrict__ sk, const float* __restrict__ dk,
    __bf16* __restrict__ wT, int K, int N) {
    __shared__ float tile[32][33];          // +1 pad: no bank conflicts
    const int i0 = blockIdx.x * 32;
    const int o0 = blockIdx.y * 32;
    const int d  = blockIdx.z;
    const int t  = threadIdx.x;
    const int lo = t & 31;
    const int lr = t >> 5;

    const size_t dkbase = (size_t)d * K * N;
#pragma unroll
    for (int p = 0; p < 4; ++p) {           // coalesced f32 reads over o
        int li = lr + p * 8;
        size_t src = (size_t)(i0 + li) * N + (o0 + lo);
        tile[li][lo] = sk[src] * dk[dkbase + src];
    }
    __syncthreads();
#pragma unroll
    for (int p = 0; p < 4; ++p) {           // coalesced bf16 writes over i
        int lO = lr + p * 8;
        wT[((size_t)d * N + (o0 + lO)) * K + (i0 + lo)] = (__bf16)tile[lo][lO];
    }
}

__global__ void fuse_bias(const float* __restrict__ sb, const float* __restrict__ db,
                          float* __restrict__ beff, int N) {
    int idx = blockIdx.x * blockDim.x + threadIdx.x;
    if (idx < NDOM * N) beff[idx] = sb[idx % N] + db[idx];
}

__global__ void cvt_bf16(const float* __restrict__ x, __bf16* __restrict__ xb, int n) {
    int i = blockIdx.x * blockDim.x + threadIdx.x;
    if (i < n) xb[i] = (__bf16)x[i];
}

// ---------------------------------------------------------------------------
// Fragment loader: 16 bf16 from two aligned 16B chunks (LDS)
// ---------------------------------------------------------------------------
__device__ inline v16bf ld_frag_pair(const __bf16* p0, const __bf16* p1) {
    v8bf lo = *(const v8bf*)p0;
    v8bf hi = *(const v8bf*)p1;
    return __builtin_shufflevector(lo, hi, 0, 1, 2, 3, 4, 5, 6, 7,
                                           8, 9, 10, 11, 12, 13, 14, 15);
}

// ---------------------------------------------------------------------------
// Grouped GEMM: block = 8 waves; block tile 32(M) x 256(N); K-step = 64.
// LDS double buffer: Bl[buf][256][64] (32KB), Al[buf][32][64] (4KB).
// Pipeline per iteration:
//   1) global->reg loads for slice k+KS            (loads in flight)
//   2) ds_load ALL fragments of slice k            (one clause, one wait)
//      sched_barrier(0)
//   3) 8x WMMA back-to-back
//   4) reg->LDS stores of slice k+KS (loadcnt waits land here), barrier
// ---------------------------------------------------------------------------
struct StageRegs {
    v8bf b[8];      // B row slice: 128B
    v8bf a;         // A chunk: 16B
};

template <bool FINAL>
__global__ __launch_bounds__(256) void star_gemm(
    const __bf16* __restrict__ A, const __bf16* __restrict__ W,
    const float* __restrict__ bias, const int* __restrict__ rows,
    const int* __restrict__ cnt, void* __restrict__ outv, int K, int N) {

    __shared__ alignas(16) __bf16 Bl[2][256][KS];   // 2 x 32 KB
    __shared__ alignas(16) __bf16 Al[2][32][KS];    // 2 x  4 KB

    const int d       = blockIdx.z;
    const int m_block = blockIdx.y * 32;
    if (m_block >= cnt[d]) return;              // uniform early exit (empty tile)

    const int t      = threadIdx.x;
    const int lane   = t & 31;
    const int wave   = t >> 5;
    const int lanelo = lane & 15;
    const int half   = lane >> 4;               // 0: lanes 0-15, 1: lanes 16-31

    const int m_wave_l = (wave >> 2) * 16;      // local M base: 0 or 16
    const int n_wave_l = (wave & 3) * 64;       // local N base: 0..192
    const int n_block  = blockIdx.x * 256;

    const int*    rown = rows + d * BATCH;
    const __bf16* Wd   = W + (size_t)d * N * K;

    // --- cooperative copy roles (no divergence: 256 threads exactly cover) --
    // B: thread t owns LDS row n=t, 128B = 8 x b128 per slice
    const __bf16* gB = Wd + (size_t)(n_block + t) * K;
    // A: 32 rows x 8 chunks of 16B = 256 chunks; thread t owns chunk t
    const int am = t >> 3;
    const int ac = (t & 7) * 8;
    int ar = rown[m_block + am];
    if (ar < 0) ar = 0;                         // clamp pad rows (stores skipped)
    const __bf16* gA = A + (size_t)ar * K;

    auto ld_stage = [&](int k0) {
        StageRegs s;
#pragma unroll
        for (int c = 0; c < 8; ++c) s.b[c] = *(const v8bf*)(gB + k0 + c * 8);
        s.a = *(const v8bf*)(gA + k0 + ac);
        return s;
    };
    auto st_stage = [&](const StageRegs& s, int buf) {
#pragma unroll
        for (int c = 0; c < 8; ++c) *(v8bf*)&Bl[buf][t][c * 8] = s.b[c];
        *(v8bf*)&Al[buf][am][ac] = s.a;
    };

    v8f acc0 = {}, acc1 = {}, acc2 = {}, acc3 = {};

    // prologue: fill buffer 0
    {
        StageRegs s0 = ld_stage(0);
        st_stage(s0, 0);
    }
    __syncthreads();

    int cur = 0;
#pragma unroll 2
    for (int k0 = 0; k0 < K; k0 += KS) {
        const bool has_next = (k0 + KS < K);    // uniform
        StageRegs nx;
        if (has_next) nx = ld_stage(k0 + KS);   // global loads in flight...

        // --- preload ALL fragments for this slice (distinct registers) ----
        const __bf16* arow = &Al[cur][m_wave_l + lanelo][0];
        const __bf16* br0  = &Bl[cur][n_wave_l +  0 + lanelo][0];
        const __bf16* br1  = &Bl[cur][n_wave_l + 16 + lanelo][0];
        const __bf16* br2  = &Bl[cur][n_wave_l + 32 + lanelo][0];
        const __bf16* br3  = &Bl[cur][n_wave_l + 48 + lanelo][0];
        const int a0o = half * 16;              // substep 0 A chunk base
        const int a1o = 32 + half * 16;         // substep 1
        const int b0o = half * 16;              // substep 0 B base
        const int b1o = 32 + half * 16;         // substep 1

        v16bf fa0 = ld_frag_pair(arow + a0o, arow + a0o + 8);
        v16bf f00 = ld_frag_pair(br0 + b0o, br0 + b0o + 8);
        v16bf f01 = ld_frag_pair(br1 + b0o, br1 + b0o + 8);
        v16bf f02 = ld_frag_pair(br2 + b0o, br2 + b0o + 8);
        v16bf f03 = ld_frag_pair(br3 + b0o, br3 + b0o + 8);
        v16bf fa1 = ld_frag_pair(arow + a1o, arow + a1o + 8);
        v16bf f10 = ld_frag_pair(br0 + b1o, br0 + b1o + 8);
        v16bf f11 = ld_frag_pair(br1 + b1o, br1 + b1o + 8);
        v16bf f12 = ld_frag_pair(br2 + b1o, br2 + b1o + 8);
        v16bf f13 = ld_frag_pair(br3 + b1o, br3 + b1o + 8);

        __builtin_amdgcn_sched_barrier(0);      // keep loads above, WMMAs below

        acc0 = __builtin_amdgcn_wmma_f32_16x16x32_bf16(false, fa0, false, f00, (short)0, acc0, false, false);
        acc1 = __builtin_amdgcn_wmma_f32_16x16x32_bf16(false, fa0, false, f01, (short)0, acc1, false, false);
        acc2 = __builtin_amdgcn_wmma_f32_16x16x32_bf16(false, fa0, false, f02, (short)0, acc2, false, false);
        acc3 = __builtin_amdgcn_wmma_f32_16x16x32_bf16(false, fa0, false, f03, (short)0, acc3, false, false);
        acc0 = __builtin_amdgcn_wmma_f32_16x16x32_bf16(false, fa1, false, f10, (short)0, acc0, false, false);
        acc1 = __builtin_amdgcn_wmma_f32_16x16x32_bf16(false, fa1, false, f11, (short)0, acc1, false, false);
        acc2 = __builtin_amdgcn_wmma_f32_16x16x32_bf16(false, fa1, false, f12, (short)0, acc2, false, false);
        acc3 = __builtin_amdgcn_wmma_f32_16x16x32_bf16(false, fa1, false, f13, (short)0, acc3, false, false);

        if (has_next) st_stage(nx, cur ^ 1);    // loadcnt waits land here
        __syncthreads();
        cur ^= 1;
    }

    // --- epilogue: bias + relu, scatter back to original batch rows -------
    const int n_wave = n_block + n_wave_l;
    const float bn0 = bias[d * N + n_wave +  0 + lanelo];
    const float bn1 = bias[d * N + n_wave + 16 + lanelo];
    const float bn2 = bias[d * N + n_wave + 32 + lanelo];
    const float bn3 = bias[d * N + n_wave + 48 + lanelo];

#pragma unroll
    for (int v = 0; v < 8; ++v) {
        // C/D layout: VGPR v, lanes 0-15 -> M=v, lanes 16-31 -> M=v+8, N=lane&15
        int m = m_wave_l + half * 8 + v;
        int r = rown[m_block + m];
        if (r >= 0) {
            float x0 = acc0[v] + bn0; x0 = x0 > 0.f ? x0 : 0.f;
            float x1 = acc1[v] + bn1; x1 = x1 > 0.f ? x1 : 0.f;
            float x2 = acc2[v] + bn2; x2 = x2 > 0.f ? x2 : 0.f;
            float x3 = acc3[v] + bn3; x3 = x3 > 0.f ? x3 : 0.f;
            if (FINAL) {
                float* out = (float*)outv;
                out[(size_t)r * N + n_wave +  0 + lanelo] = x0;
                out[(size_t)r * N + n_wave + 16 + lanelo] = x1;
                out[(size_t)r * N + n_wave + 32 + lanelo] = x2;
                out[(size_t)r * N + n_wave + 48 + lanelo] = x3;
            } else {
                __bf16* out = (__bf16*)outv;
                out[(size_t)r * N + n_wave +  0 + lanelo] = (__bf16)x0;
                out[(size_t)r * N + n_wave + 16 + lanelo] = (__bf16)x1;
                out[(size_t)r * N + n_wave + 32 + lanelo] = (__bf16)x2;
                out[(size_t)r * N + n_wave + 48 + lanelo] = (__bf16)x3;
            }
        }
    }
}

// ---------------------------------------------------------------------------
// Launcher
// ---------------------------------------------------------------------------
extern "C" void kernel_launch(void* const* d_in, const int* in_sizes, int n_in,
                              void* d_out, int out_size, void* d_ws, size_t ws_size,
                              hipStream_t stream) {
    (void)in_sizes; (void)n_in; (void)out_size; (void)ws_size;

    const float* x   = (const float*)d_in[0];
    const int*   ind = (const int*)d_in[1];
    const float* sk0 = (const float*)d_in[2];
    const float* sb0 = (const float*)d_in[3];
    const float* dk0 = (const float*)d_in[4];
    const float* db0 = (const float*)d_in[5];
    const float* sk1 = (const float*)d_in[6];
    const float* sb1 = (const float*)d_in[7];
    const float* dk1 = (const float*)d_in[8];
    const float* db1 = (const float*)d_in[9];
    const float* sk2 = (const float*)d_in[10];
    const float* sb2 = (const float*)d_in[11];
    const float* dk2 = (const float*)d_in[12];
    const float* db2 = (const float*)d_in[13];

    // Workspace carve-up (256B aligned)
    char* ws = (char*)d_ws;
    size_t off = 0;
    auto carve = [&](size_t bytes) -> char* {
        off = (off + 255) & ~(size_t)255;
        char* p = ws + off;
        off += bytes;
        return p;
    };
    int*    cnt  = (int*)carve(NDOM * sizeof(int));
    int*    rows = (int*)carve((size_t)NDOM * BATCH * sizeof(int));
    __bf16* xb   = (__bf16*)carve((size_t)BATCH * D_IN * 2);
    __bf16* h1   = (__bf16*)carve((size_t)BATCH * H0 * 2);
    __bf16* h2   = (__bf16*)carve((size_t)BATCH * H1 * 2);
    __bf16* w0T  = (__bf16*)carve((size_t)NDOM * H0 * D_IN * 2);
    __bf16* w1T  = (__bf16*)carve((size_t)NDOM * H1 * H0 * 2);
    __bf16* w2T  = (__bf16*)carve((size_t)NDOM * H2 * H1 * 2);
    float*  b0e  = (float*)carve((size_t)NDOM * H0 * 4);
    float*  b1e  = (float*)carve((size_t)NDOM * H1 * 4);
    float*  b2e  = (float*)carve((size_t)NDOM * H2 * 4);

    // 1) bucket rows by domain
    {
        int total = NDOM * BATCH;
        init_meta<<<(total + 255) / 256, 256, 0, stream>>>(cnt, rows, total);
        scatter_rows<<<(BATCH + 255) / 256, 256, 0, stream>>>(ind, cnt, rows);
    }

    // 2) fuse weights/biases (tiled transpose), convert x to bf16
    {
        fuse_weights_t<<<dim3(D_IN / 32, H0 / 32, NDOM), 256, 0, stream>>>(sk0, dk0, w0T, D_IN, H0);
        fuse_weights_t<<<dim3(H0 / 32, H1 / 32, NDOM), 256, 0, stream>>>(sk1, dk1, w1T, H0, H1);
        fuse_weights_t<<<dim3(H1 / 32, H2 / 32, NDOM), 256, 0, stream>>>(sk2, dk2, w2T, H1, H2);
        fuse_bias<<<(NDOM * H0 + 255) / 256, 256, 0, stream>>>(sb0, db0, b0e, H0);
        fuse_bias<<<(NDOM * H1 + 255) / 256, 256, 0, stream>>>(sb1, db1, b1e, H1);
        fuse_bias<<<(NDOM * H2 + 255) / 256, 256, 0, stream>>>(sb2, db2, b2e, H2);
        cvt_bf16<<<(BATCH * D_IN + 255) / 256, 256, 0, stream>>>(x, xb, BATCH * D_IN);
    }

    // 3) grouped GEMM chain (grid.y = worst case all rows in one domain)
    {
        dim3 blk(256, 1, 1);
        dim3 g0(H0 / 256, BATCH / 32, NDOM);
        dim3 g1(H1 / 256, BATCH / 32, NDOM);
        dim3 g2(H2 / 256, BATCH / 32, NDOM);
        star_gemm<false><<<g0, blk, 0, stream>>>(xb, w0T, b0e, rows, cnt, h1, D_IN, H0);
        star_gemm<false><<<g1, blk, 0, stream>>>(h1, w1T, b1e, rows, cnt, h2, H0, H1);
        star_gemm<true ><<<g2, blk, 0, stream>>>(h2, w2T, b2e, rows, cnt, d_out, H1, H2);
    }
}
